// LatentActionQuantization_67001489817948
// MI455X (gfx1250) — matrix operand order repaired and checked by reading, not compiled, not run on hardware
//
#include <hip/hip_runtime.h>
#include <math.h>

// ---------------- static model dims ----------------
#define B_    8
#define C_    3
#define T_    2
#define H_    256
#define W_    256
#define P_    16
#define GH_   16
#define GW_   16
#define HW_   256
#define D_    512
#define HEADS_ 8
#define DH_   64
#define FF_   2048
#define Q_    256
#define KCB_  1024
#define CPP_  768
#define NTOK_ 4096           // B*T*HW == B*HW*T
#define SCALE_ 0.125f        // 1/sqrt(64)

typedef __attribute__((ext_vector_type(16))) __bf16 v16bf;
typedef __attribute__((ext_vector_type(8)))  __bf16 v8bf;
typedef __attribute__((ext_vector_type(4)))  __bf16 v4bf;
typedef __attribute__((ext_vector_type(8)))  float  v8f;
typedef __attribute__((ext_vector_type(4)))  float  f4;

__device__ __forceinline__ float gelu_f(float x) {
  const float c = 0.7978845608028654f;   // jax.nn.gelu approximate=True
  return 0.5f * x * (1.0f + tanhf(c * (x + 0.044715f * x * x * x)));
}

__device__ __forceinline__ v16bf cat8(v8bf lo, v8bf hi) {
  return __builtin_shufflevector(lo, hi, 0,1,2,3,4,5,6,7,8,9,10,11,12,13,14,15);
}
__device__ __forceinline__ v4bf cvt4(f4 v) {
  v4bf r; r[0] = (__bf16)v[0]; r[1] = (__bf16)v[1]; r[2] = (__bf16)v[2]; r[3] = (__bf16)v[3];
  return r;
}
#define WMMA_BF16(a, b, c) \
  __builtin_amdgcn_wmma_f32_16x16x32_bf16(false, (a), false, (b), (short)0, (c), false, false)

// ======================================================================
// Guard-free GEMM: C[M,N] = act( A[M,K] @ B[K,N] + bias ), bf16 WMMA,
// f32 acc. REQUIRES M%64==0, N%64==0, K%64==0.
// 256 thr = 8 waves; 64x64 tile; wave (wm=wid>>1, wn=wid&1) owns 16x32.
// Double-buffered LDS (ping-pong), batched f4 global loads, one barrier
// per K-step. LDS row stride 72 elems -> fragments = aligned ds_load_b128.
// ======================================================================
#define GT_M 64
#define GT_N 64
#define GT_K 64
#define LSTR 72

__global__ __launch_bounds__(256)
void gemm_bf16_wmma(const float* __restrict__ A, const float* __restrict__ Bm,
                    const float* __restrict__ bias, float* __restrict__ C,
                    int M, int N, int K, int lda, int ldb, int ldc, int act)
{
  (void)M;
  __shared__ __attribute__((aligned(16))) __bf16 As [2][GT_M][LSTR];
  __shared__ __attribute__((aligned(16))) __bf16 BsT[2][GT_N][LSTR];
  const int tid  = threadIdx.x;
  const int lane = tid & 31, wid = tid >> 5;
  const int wm = wid >> 1, wn = wid & 1;
  const int row0 = blockIdx.y * GT_M;
  const int col0 = blockIdx.x * GT_N;

  // per-thread staging coords (both tiles read row-major as f4)
  int rr[4], cc[4];
#pragma unroll
  for (int j = 0; j < 4; ++j) {
    int idx4 = tid + j * 256;            // 1024 f4 per 64x64 tile
    rr[j] = idx4 >> 4;                   // row (m for A, k for B)
    cc[j] = (idx4 & 15) << 2;            // col offset (k for A, n for B)
  }

  auto stage = [&](int k0, int p) {
    f4 va[4], vb[4];
#pragma unroll
    for (int j = 0; j < 4; ++j)
      va[j] = *(const f4*)(A + (row0 + rr[j]) * lda + k0 + cc[j]);
#pragma unroll
    for (int j = 0; j < 4; ++j)
      vb[j] = *(const f4*)(Bm + (k0 + rr[j]) * ldb + col0 + cc[j]);
#pragma unroll
    for (int j = 0; j < 4; ++j)
      *(v4bf*)&As[p][rr[j]][cc[j]] = cvt4(va[j]);
#pragma unroll
    for (int j = 0; j < 4; ++j) {
#pragma unroll
      for (int i = 0; i < 4; ++i)         // transposed scatter: BsT[n][k]
        BsT[p][cc[j] + i][rr[j]] = (__bf16)vb[j][i];
    }
  };

  v8f acc0 = {}; v8f acc1 = {};
  const int ml   = wm * 16 + (lane & 15);
  const int nl   = wn * 32 + (lane & 15);
  const int kh8  = (lane >> 4) << 3;
  const int kh16 = (lane >> 4) << 4;

  stage(0, 0);
  __syncthreads();
  int p = 0;
  for (int k0 = 0; k0 < K; k0 += GT_K) {
    if (k0 + GT_K < K) stage(k0 + GT_K, p ^ 1);
    if (k0 + 2 * GT_K < K)               // global_prefetch_b8, 2 tiles ahead
      __builtin_prefetch(A + (row0 + (tid & 63)) * lda + k0 + 2 * GT_K, 0, 1);
#pragma unroll
    for (int ks = 0; ks < GT_K; ks += 32) {
      v16bf af  = cat8(*(const v8bf*)&As [p][ml][ks + kh16],
                       *(const v8bf*)&As [p][ml][ks + kh16 + 8]);
      v16bf bf0 = cat8(*(const v8bf*)&BsT[p][nl][ks + kh16],
                       *(const v8bf*)&BsT[p][nl][ks + kh16 + 8]);
      v16bf bf1 = cat8(*(const v8bf*)&BsT[p][nl + 16][ks + kh16],
                       *(const v8bf*)&BsT[p][nl + 16][ks + kh16 + 8]);
      acc0 = WMMA_BF16(af, bf0, acc0);
      acc1 = WMMA_BF16(af, bf1, acc1);
    }
    __syncthreads();
    p ^= 1;
  }

  const int mofs = (lane >> 4) << 3;
  const int ncol = lane & 15;
#pragma unroll
  for (int r = 0; r < 8; ++r) {
    int m  = row0 + wm * 16 + r + mofs;
    int n0 = col0 + wn * 32 + ncol;
    int n1 = n0 + 16;
    float v0 = acc0[r] + (bias ? bias[n0] : 0.0f);
    float v1 = acc1[r] + (bias ? bias[n1] : 0.0f);
    if (act == 1) { v0 = gelu_f(v0); v1 = gelu_f(v1); }
    C[m * ldc + n0] = v0;
    C[m * ldc + n1] = v1;
  }
}

// Tiny GEMM for M=8 bottleneck matmuls (fp32, negligible FLOPs)
__global__ __launch_bounds__(256)
void gemm_small(const float* __restrict__ A, const float* __restrict__ Bm,
                const float* __restrict__ bias, float* __restrict__ C,
                int M, int N, int K, int lda, int ldb, int ldc)
{
  int id = blockIdx.x * blockDim.x + threadIdx.x;
  if (id >= M * N) return;
  int m = id / N, n = id % N;
  float acc = 0.f;
  for (int k = 0; k < K; ++k) acc += A[m * lda + k] * Bm[k * ldb + n];
  C[m * ldc + n] = acc + (bias ? bias[n] : 0.0f);
}

// ======================================================================
// Fused self-attention S=256, DH=64; qkv rows [(n,s),1536] = q|k|v (h,dh)
// grid: (S/32, NB*HEADS). All WMMA fragments are aligned b128 LDS loads.
// ======================================================================
#define AROWS 32

__global__ __launch_bounds__(256)
void attn_fused(const float* __restrict__ qkv, const float* __restrict__ biasb,
                float* __restrict__ out)
{
  __shared__ __attribute__((aligned(16))) float  sc [AROWS][HW_]; // 32 KB scores
  __shared__ __attribute__((aligned(16))) __bf16 Pb [AROWS][HW_]; // 16 KB probs
  __shared__ __attribute__((aligned(16))) __bf16 Aq [AROWS][72];  // q tile
  __shared__ __attribute__((aligned(16))) __bf16 Bsh[64][72];     // K^T / V tile

  const int tid  = threadIdx.x;
  const int lane = tid & 31, wid = tid >> 5;
  const int wm = wid >> 2;                  // 0..1
  const int wn = wid & 3;                   // 0..3
  const int n = blockIdx.y / HEADS_;
  const int h = blockIdx.y % HEADS_;
  const int sbase = blockIdx.x * AROWS;
  const int LD = 3 * D_;
  const int ml   = wm * 16 + (lane & 15);
  const int nl   = wn * 16 + (lane & 15);
  const int kh8  = (lane >> 4) << 3;
  const int kh16 = (lane >> 4) << 4;
  const int mofs = (lane >> 4) << 3;

  int rr[4], cc[4];
#pragma unroll
  for (int j = 0; j < 4; ++j) {
    int idx4 = tid + j * 256;
    rr[j] = idx4 >> 4;                     // row within 64-row tile
    cc[j] = (idx4 & 15) << 2;              // f4 column offset within 64
  }

  // stage 0: q tile (32x64) -> LDS bf16 (batched loads)
  {
    f4 vq[2];
#pragma unroll
    for (int j = 0; j < 2; ++j)
      vq[j] = *(const f4*)(qkv + (n * HW_ + sbase + rr[j]) * LD + h * DH_ + cc[j]);
#pragma unroll
    for (int j = 0; j < 2; ++j)
      *(v4bf*)&Aq[rr[j]][cc[j]] = cvt4(vq[j]);
  }
  __syncthreads();

  // stage 1: scores = q @ k^T * scale + bias, in 64-key chunks
  for (int c = 0; c < 4; ++c) {
    {
      f4 vk[4];
#pragma unroll
      for (int j = 0; j < 4; ++j)          // Kt[key][feature], row-major loads
        vk[j] = *(const f4*)(qkv + (n * HW_ + c * 64 + rr[j]) * LD + D_ + h * DH_ + cc[j]);
#pragma unroll
      for (int j = 0; j < 4; ++j)
        *(v4bf*)&Bsh[rr[j]][cc[j]] = cvt4(vk[j]);
    }
    __syncthreads();

    v8f acc = {};
#pragma unroll
    for (int ks = 0; ks < 64; ks += 32) {
      v16bf af = cat8(*(const v8bf*)&Aq [ml][ks + kh16], *(const v8bf*)&Aq [ml][ks + kh16 + 8]);
      v16bf bf = cat8(*(const v8bf*)&Bsh[nl][ks + kh16], *(const v8bf*)&Bsh[nl][ks + kh16 + 8]);
      acc = WMMA_BF16(af, bf, acc);
    }
#pragma unroll
    for (int r = 0; r < 8; ++r) {
      int m  = wm * 16 + r + mofs;
      int jg = c * 64 + wn * 16 + (lane & 15);
      sc[m][jg] = acc[r] * SCALE_ + biasb[h * HW_ * HW_ + (sbase + m) * HW_ + jg];
    }
    __syncthreads();
  }

  // stage 2: softmax, 8 threads per row (wave32 shuffles), emit bf16 probs
  {
    int row = tid >> 3, sub = tid & 7;
    f4* scr = (f4*)&sc[row][0];
    float mx = -3.0e38f;
#pragma unroll
    for (int j4 = sub * 8; j4 < sub * 8 + 8; ++j4) {
      f4 v = scr[j4];
      mx = fmaxf(mx, fmaxf(fmaxf(v[0], v[1]), fmaxf(v[2], v[3])));
    }
    for (int m = 1; m < 8; m <<= 1) mx = fmaxf(mx, __shfl_xor(mx, m, 32));
    float s = 0.0f;
#pragma unroll
    for (int j4 = sub * 8; j4 < sub * 8 + 8; ++j4) {
      f4 v = scr[j4];
      v[0] = expf(v[0] - mx); v[1] = expf(v[1] - mx);
      v[2] = expf(v[2] - mx); v[3] = expf(v[3] - mx);
      scr[j4] = v;
      s += v[0] + v[1] + v[2] + v[3];
    }
    for (int m = 1; m < 8; m <<= 1) s += __shfl_xor(s, m, 32);
    float inv = 1.0f / s;
#pragma unroll
    for (int j4 = sub * 8; j4 < sub * 8 + 8; ++j4) {
      f4 v = scr[j4];
      v[0] *= inv; v[1] *= inv; v[2] *= inv; v[3] *= inv;
      *(v4bf*)&Pb[row][j4 << 2] = cvt4(v);
    }
  }

  // stage 3: o = probs @ v, 64-key chunks (K=256)
  v8f oacc = {};
  for (int ks = 0; ks < 4; ++ks) {
    __syncthreads();
    {
      f4 vv[4];
#pragma unroll
      for (int j = 0; j < 4; ++j)          // V row-major loads (key, valdim)
        vv[j] = *(const f4*)(qkv + (n * HW_ + ks * 64 + rr[j]) * LD + 2 * D_ + h * DH_ + cc[j]);
#pragma unroll
      for (int j = 0; j < 4; ++j) {
#pragma unroll
        for (int i = 0; i < 4; ++i)        // transposed scatter: Vt[valdim][key]
          Bsh[cc[j] + i][rr[j]] = (__bf16)vv[j][i];
      }
    }
    __syncthreads();
#pragma unroll
    for (int k2 = 0; k2 < 64; k2 += 32) {
      int kb = ks * 64 + k2;
      v16bf af = cat8(*(const v8bf*)&Pb [ml][kb + kh16], *(const v8bf*)&Pb [ml][kb + kh16 + 8]);
      v16bf bf = cat8(*(const v8bf*)&Bsh[nl][k2 + kh16], *(const v8bf*)&Bsh[nl][k2 + kh16 + 8]);
      oacc = WMMA_BF16(af, bf, oacc);
    }
  }
#pragma unroll
  for (int r = 0; r < 8; ++r) {
    int m = sbase + wm * 16 + r + mofs;
    int col = h * DH_ + wn * 16 + (lane & 15);
    out[(n * HW_ + m) * D_ + col] = oacc[r];
  }
}

// Temporal attention, T=2: one thread per (n, h, i), float4 math
__global__ __launch_bounds__(256)
void attn_t2(const float* __restrict__ qkv, float* __restrict__ out)
{
  int id = blockIdx.x * blockDim.x + threadIdx.x;
  if (id >= B_ * HW_ * HEADS_ * T_) return;
  int i = id & 1, h = (id >> 1) & 7, n = id >> 4;
  const int LD = 3 * D_;
  const f4* qp = (const f4*)(qkv + (n * 2 + i) * LD + h * DH_);
  const f4* k0 = (const f4*)(qkv + (n * 2 + 0) * LD + D_ + h * DH_);
  const f4* k1 = (const f4*)(qkv + (n * 2 + 1) * LD + D_ + h * DH_);
  const f4* v0 = (const f4*)(qkv + (n * 2 + 0) * LD + 2 * D_ + h * DH_);
  const f4* v1 = (const f4*)(qkv + (n * 2 + 1) * LD + 2 * D_ + h * DH_);
  float s0 = 0.f, s1 = 0.f;
#pragma unroll
  for (int d = 0; d < DH_ / 4; ++d) {
    f4 q = qp[d], a = k0[d], b = k1[d];
    s0 += q[0] * a[0] + q[1] * a[1] + q[2] * a[2] + q[3] * a[3];
    s1 += q[0] * b[0] + q[1] * b[1] + q[2] * b[2] + q[3] * b[3];
  }
  s0 *= SCALE_; s1 *= SCALE_;
  float mx = fmaxf(s0, s1);
  float e0 = expf(s0 - mx), e1 = expf(s1 - mx);
  float inv = 1.0f / (e0 + e1);
  float a0 = e0 * inv, a1 = e1 * inv;
  f4* op = (f4*)(out + (n * 2 + i) * D_ + h * DH_);
#pragma unroll
  for (int d = 0; d < DH_ / 4; ++d) op[d] = v0[d] * a0 + v1[d] * a1;
}

// ---------------- LayerNorm (block per row, float4) ----------------
__global__ __launch_bounds__(256)
void layernorm_k(const float* __restrict__ x, const float* __restrict__ g,
                 const float* __restrict__ b, float* __restrict__ y, int dim)
{
  __shared__ float red[256];
  const int row = blockIdx.x, tid = threadIdx.x;
  const int dim4 = dim >> 2;
  const f4* xr = (const f4*)(x + row * dim);
  float s = 0.f;
  for (int i = tid; i < dim4; i += 256) { f4 v = xr[i]; s += v[0] + v[1] + v[2] + v[3]; }
  red[tid] = s; __syncthreads();
  for (int off = 128; off > 0; off >>= 1) { if (tid < off) red[tid] += red[tid + off]; __syncthreads(); }
  float mean = red[0] / dim; __syncthreads();
  float vv = 0.f;
  for (int i = tid; i < dim4; i += 256) {
    f4 v = xr[i];
    float d0 = v[0] - mean, d1 = v[1] - mean, d2 = v[2] - mean, d3 = v[3] - mean;
    vv += d0 * d0 + d1 * d1 + d2 * d2 + d3 * d3;
  }
  red[tid] = vv; __syncthreads();
  for (int off = 128; off > 0; off >>= 1) { if (tid < off) red[tid] += red[tid + off]; __syncthreads(); }
  float inv = rsqrtf(red[0] / dim + 1e-5f);
  const f4* g4 = (const f4*)g; const f4* b4 = (const f4*)b;
  f4* y4 = (f4*)(y + row * dim);
  for (int i = tid; i < dim4; i += 256) {
    f4 v = xr[i], gg = g4[i], bb = b4[i], o;
    o[0] = (v[0] - mean) * inv * gg[0] + bb[0];
    o[1] = (v[1] - mean) * inv * gg[1] + bb[1];
    o[2] = (v[2] - mean) * inv * gg[2] + bb[2];
    o[3] = (v[3] - mean) * inv * gg[3] + bb[3];
    y4[i] = o;
  }
}

__global__ __launch_bounds__(256)
void add4_k(float* __restrict__ dst, const float* __restrict__ src, int n4)
{
  int i = blockIdx.x * blockDim.x + threadIdx.x;
  if (i < n4) { f4 a = ((const f4*)dst)[i], b = ((const f4*)src)[i]; ((f4*)dst)[i] = a + b; }
}

// video [B,C,T,H,W] -> xraw [(b,t,gh,gw), CPP], 4 pixels (along W) per thread
__global__ __launch_bounds__(256)
void patch_gather(const float* __restrict__ video, float* __restrict__ out)
{
  int idx4 = blockIdx.x * blockDim.x + threadIdx.x;
  if (idx4 >= NTOK_ * CPP_ / 4) return;
  int row = idx4 / (CPP_ / 4), cpp = (idx4 % (CPP_ / 4)) << 2;
  int b = row >> 9, t = (row >> 8) & 1, gh = (row >> 4) & 15, gw = row & 15;
  int c = cpp >> 8, p1 = (cpp >> 4) & 15, p2 = cpp & 15;   // p2 in {0,4,8,12}
  int vi = (((b * C_ + c) * T_ + t) * H_ + (gh * 16 + p1)) * W_ + (gw * 16 + p2);
  ((f4*)out)[idx4] = *(const f4*)(video + vi);
}

// pix [(b,t,gh,gw), CPP] -> recon [B,C,T,H,W], 4 pixels (along W) per thread
__global__ __launch_bounds__(256)
void scatter_pix(const float* __restrict__ pix, float* __restrict__ out)
{
  int idx4 = blockIdx.x * blockDim.x + threadIdx.x;
  if (idx4 >= B_ * C_ * T_ * H_ * W_ / 4) return;
  int idx = idx4 << 2;
  int b = idx / (C_ * T_ * H_ * W_);
  int r = idx % (C_ * T_ * H_ * W_);
  int c = r / (T_ * H_ * W_); r %= (T_ * H_ * W_);
  int t = r / (H_ * W_); r %= (H_ * W_);
  int hh = r / W_, ww = r % W_;
  int gh = hh >> 4, p1 = hh & 15, gw = ww >> 4, p2 = ww & 15;
  int row = ((b * T_ + t) * GH_ + gh) * GW_ + gw;
  int cpp = c * 256 + p1 * 16 + p2;
  ((f4*)out)[idx4] = *(const f4*)(pix + row * CPP_ + cpp);
}

// continuous relative position bias table -> [HEADS, HW, HW]
__global__ __launch_bounds__(256)
void cpb_k(const float* __restrict__ w1, const float* __restrict__ b1,
           const float* __restrict__ w2, const float* __restrict__ b2,
           float* __restrict__ biasb)
{
  int i = blockIdx.x, j = threadIdx.x;
  float dy = (float)((i >> 4) - (j >> 4));
  float dx = (float)((i & 15) - (j & 15));
  float acc[HEADS_];
#pragma unroll
  for (int h = 0; h < HEADS_; ++h) acc[h] = 0.f;
  for (int d = 0; d < D_; ++d) {
    float hv = fmaxf(dy * w1[d] + dx * w1[D_ + d] + b1[d], 0.f);
#pragma unroll
    for (int h = 0; h < HEADS_; ++h) acc[h] += hv * w2[d * HEADS_ + h];
  }
#pragma unroll
  for (int h = 0; h < HEADS_; ++h)
    biasb[h * HW_ * HW_ + i * HW_ + j] = acc[h] + b2[h];
}

// tok [(b,t,hw),D] <-> tm [(b,hw,t),D]  (float4 along D)
__global__ __launch_bounds__(256)
void bt_to_tm(const float* __restrict__ in, float* __restrict__ out)
{
  int idx4 = blockIdx.x * blockDim.x + threadIdx.x;
  if (idx4 >= NTOK_ * D_ / 4) return;
  int srow = idx4 >> 7, d4 = idx4 & 127;
  int b = srow >> 9, t = (srow >> 8) & 1, hw = srow & 255;
  int drow = ((b << 8) + hw) * 2 + t;
  ((f4*)out)[drow * 128 + d4] = ((const f4*)in)[idx4];
}

__global__ __launch_bounds__(256)
void tm_to_bt(const float* __restrict__ in, float* __restrict__ out)
{
  int idx4 = blockIdx.x * blockDim.x + threadIdx.x;
  if (idx4 >= NTOK_ * D_ / 4) return;
  int drow = idx4 >> 7, d4 = idx4 & 127;
  int b = drow >> 9, t = (drow >> 8) & 1, hw = drow & 255;
  int srow = ((b << 8) + hw) * 2 + t;
  ((f4*)out)[idx4] = ((const f4*)in)[srow * 128 + d4];
}

// feat[b,d] = mean_{hw} tok[(b, t=1, hw), d]
__global__ __launch_bounds__(256)
void mean_k(const float* __restrict__ tok, float* __restrict__ feat)
{
  int id4 = blockIdx.x * blockDim.x + threadIdx.x;
  if (id4 >= B_ * D_ / 4) return;
  int b = id4 >> 7, d4 = id4 & 127;
  f4 s = {};
  for (int hw = 0; hw < HW_; ++hw)
    s += ((const f4*)tok)[((b * T_ + 1) * HW_ + hw) * 128 + d4];
  ((f4*)feat)[id4] = s * (1.0f / HW_);
}

__global__ __launch_bounds__(256)
void vq_argmin(const float* __restrict__ z, const float* __restrict__ cb, int* __restrict__ idx)
{
  __shared__ float zrow[Q_];
  __shared__ float sd[256];
  __shared__ int   si[256];
  int b = blockIdx.x, tid = threadIdx.x;
  zrow[tid] = z[b * Q_ + tid];
  __syncthreads();
  float best = 3.0e38f; int bi = 0;
  for (int k = tid; k < KCB_; k += 256) {
    const f4* cr = (const f4*)(cb + k * Q_);
    float d2 = 0.f;
    for (int q4 = 0; q4 < Q_ / 4; ++q4) {
      f4 c = cr[q4];
      float f0 = zrow[q4 * 4 + 0] - c[0], f1 = zrow[q4 * 4 + 1] - c[1];
      float f2 = zrow[q4 * 4 + 2] - c[2], f3 = zrow[q4 * 4 + 3] - c[3];
      d2 += f0 * f0 + f1 * f1 + f2 * f2 + f3 * f3;
    }
    if (d2 < best) { best = d2; bi = k; }
  }
  sd[tid] = best; si[tid] = bi; __syncthreads();
  for (int off = 128; off > 0; off >>= 1) {
    if (tid < off) {
      if (sd[tid + off] < sd[tid] || (sd[tid + off] == sd[tid] && si[tid + off] < si[tid])) {
        sd[tid] = sd[tid + off]; si[tid] = si[tid + off];
      }
    }
    __syncthreads();
  }
  if (tid == 0) idx[b] = si[0];
}

__global__ __launch_bounds__(256)
void vq_nsvq(const float* __restrict__ z, const float* __restrict__ cb,
             const int* __restrict__ idx, const float* __restrict__ noise,
             float* __restrict__ zq)
{
  __shared__ float red[256];
  int b = blockIdx.x, tid = threadIdx.x;
  float zv = z[b * Q_ + tid];
  float cv = cb[idx[b] * Q_ + tid];
  float nv = noise[b * Q_ + tid];
  float dv = zv - cv;
  red[tid] = dv * dv; __syncthreads();
  for (int off = 128; off > 0; off >>= 1) { if (tid < off) red[tid] += red[tid + off]; __syncthreads(); }
  float err = sqrtf(red[0]); __syncthreads();
  red[tid] = nv * nv; __syncthreads();
  for (int off = 128; off > 0; off >>= 1) { if (tid < off) red[tid] += red[tid + off]; __syncthreads(); }
  float nn = sqrtf(red[0]);
  zq[b * Q_ + tid] = zv + err / (nn + 1e-8f) * nv;
}

// x[(b,t,s),d] += vo8[b,d]  (cross-attn with L=1: softmax == 1, out = v)
__global__ __launch_bounds__(256)
void bcast_add(float* __restrict__ x, const float* __restrict__ vo8)
{
  int idx4 = blockIdx.x * blockDim.x + threadIdx.x;
  if (idx4 >= NTOK_ * D_ / 4) return;
  int row = idx4 >> 7;
  int b = row >> 9;
  f4 a = ((const f4*)x)[idx4];
  f4 v = ((const f4*)vo8)[(b << 7) + (idx4 & 127)];
  ((f4*)x)[idx4] = a + v;
}

// ======================================================================
extern "C" void kernel_launch(void* const* d_in, const int* in_sizes, int n_in,
                              void* d_out, int out_size, void* d_ws, size_t ws_size,
                              hipStream_t stream)
{
  (void)in_sizes; (void)n_in; (void)ws_size;
  int ii = 0;
  const float* video   = (const float*)d_in[ii++];
  const float* pe_ln1g = (const float*)d_in[ii++];
  const float* pe_ln1b = (const float*)d_in[ii++];
  const float* pe_w    = (const float*)d_in[ii++];
  const float* pe_b    = (const float*)d_in[ii++];
  const float* pe_ln2g = (const float*)d_in[ii++];
  const float* pe_ln2b = (const float*)d_in[ii++];
  const float* cpb_w1  = (const float*)d_in[ii++];
  const float* cpb_b1  = (const float*)d_in[ii++];
  const float* cpb_w2  = (const float*)d_in[ii++];
  const float* cpb_b2  = (const float*)d_in[ii++];
  const float* es_ln1g = (const float*)d_in[ii++];
  const float* es_ln1b = (const float*)d_in[ii++];
  const float* es_wqkv = (const float*)d_in[ii++];
  const float* es_wo   = (const float*)d_in[ii++];
  const float* es_ln2g = (const float*)d_in[ii++];
  const float* es_ln2b = (const float*)d_in[ii++];
  const float* es_w1   = (const float*)d_in[ii++];
  const float* es_b1   = (const float*)d_in[ii++];
  const float* es_w2   = (const float*)d_in[ii++];
  const float* es_b2   = (const float*)d_in[ii++];
  const float* et_ln1g = (const float*)d_in[ii++];
  const float* et_ln1b = (const float*)d_in[ii++];
  const float* et_wqkv = (const float*)d_in[ii++];
  const float* et_wo   = (const float*)d_in[ii++];
  const float* et_ln2g = (const float*)d_in[ii++];
  const float* et_ln2b = (const float*)d_in[ii++];
  const float* et_w1   = (const float*)d_in[ii++];
  const float* et_b1   = (const float*)d_in[ii++];
  const float* et_w2   = (const float*)d_in[ii++];
  const float* et_b2   = (const float*)d_in[ii++];
  const float* ds_ln1g = (const float*)d_in[ii++];
  const float* ds_ln1b = (const float*)d_in[ii++];
  const float* ds_wqkv = (const float*)d_in[ii++];
  const float* ds_wo   = (const float*)d_in[ii++];
  const float* ds_clng = (const float*)d_in[ii++]; (void)ds_clng; // L=1 cross: q-path
  const float* ds_clnb = (const float*)d_in[ii++]; (void)ds_clnb; //   is irrelevant
  const float* ds_wq   = (const float*)d_in[ii++]; (void)ds_wq;
  const float* ds_wkv  = (const float*)d_in[ii++];
  const float* ds_cwo  = (const float*)d_in[ii++];
  const float* ds_ln2g = (const float*)d_in[ii++];
  const float* ds_ln2b = (const float*)d_in[ii++];
  const float* ds_w1   = (const float*)d_in[ii++];
  const float* ds_b1   = (const float*)d_in[ii++];
  const float* ds_w2   = (const float*)d_in[ii++];
  const float* ds_b2   = (const float*)d_in[ii++];
  const float* vq_inw  = (const float*)d_in[ii++];
  const float* vq_inb  = (const float*)d_in[ii++];
  const float* codebook= (const float*)d_in[ii++];
  const float* vq_outw = (const float*)d_in[ii++];
  const float* vq_outb = (const float*)d_in[ii++];
  const float* noise   = (const float*)d_in[ii++];
  const float* px_w    = (const float*)d_in[ii++];
  const float* px_b    = (const float*)d_in[ii++];

  // ---- workspace layout (floats); total ~25.7M floats (~103 MB) ----
  float* ws    = (float*)d_ws;
  float* R0    = ws;                     // 8,388,608  (xraw / ff1 / pix)
  float* tok   = R0    + 8388608;        // 2,097,152
  float* xn    = tok   + 2097152;        // 2,097,152
  float* qkvb  = xn    + 2097152;        // 6,291,456  (also ln768 temp)
  float* attno = qkvb  + 6291456;        // 2,097,152
  float* projb = attno + 2097152;        // 2,097,152
  float* tmb   = projb + 2097152;        // 2,097,152
  float* biasb = tmb   + 2097152;        //   524,288
  float* feat  = biasb + 524288;         //     4,096
  float* zb    = feat  + 4096;           //     2,048
  float* zqb   = zb    + 2048;           //     2,048
  float* actb  = zqb   + 2048;           //     4,096
  float* v8    = actb  + 4096;           //     4,096
  float* vo8   = v8    + 4096;           //     4,096
  int*   idxb  = (int*)(vo8 + 4096);

  auto gemm = [&](const float* A, const float* Bm, const float* bias, float* Cm,
                  int M, int N, int K, int lda, int ldb, int ldc, int act) {
    dim3 g(N / 64, M / 64);              // M,N,K multiples of 64 (guard-free)
    gemm_bf16_wmma<<<g, 256, 0, stream>>>(A, Bm, bias, Cm, M, N, K, lda, ldb, ldc, act);
  };
  auto gemm8 = [&](const float* A, const float* Bm, const float* bias, float* Cm,
                   int M, int N, int K, int lda, int ldb, int ldc) {
    gemm_small<<<(M * N + 255) / 256, 256, 0, stream>>>(A, Bm, bias, Cm, M, N, K, lda, ldb, ldc);
  };
  auto lnorm = [&](const float* x, const float* g, const float* b, float* y, int rows, int dim) {
    layernorm_k<<<rows, 256, 0, stream>>>(x, g, b, y, dim);
  };
  auto addv = [&](float* dst, const float* src, int n) {
    add4_k<<<(n / 4 + 255) / 256, 256, 0, stream>>>(dst, src, n / 4);
  };

  // ---- patch embed ----
  patch_gather<<<(NTOK_ * CPP_ / 4 + 255) / 256, 256, 0, stream>>>(video, R0);
  lnorm(R0, pe_ln1g, pe_ln1b, qkvb, NTOK_, CPP_);
  gemm(qkvb, pe_w, pe_b, projb, NTOK_, D_, CPP_, CPP_, D_, D_, 0);
  lnorm(projb, pe_ln2g, pe_ln2b, tok, NTOK_, D_);

  // ---- relative position bias ----
  cpb_k<<<HW_, HW_, 0, stream>>>(cpb_w1, cpb_b1, cpb_w2, cpb_b2, biasb);

  auto self_layer = [&](float* x, const float* l1g, const float* l1b, const float* wqkv,
                        const float* wo, const float* l2g, const float* l2b,
                        const float* w1, const float* b1, const float* w2, const float* b2,
                        int spatial) {
    lnorm(x, l1g, l1b, xn, NTOK_, D_);
    gemm(xn, wqkv, nullptr, qkvb, NTOK_, 3 * D_, D_, D_, 3 * D_, 3 * D_, 0);
    if (spatial) {
      dim3 ga(HW_ / AROWS, (B_ * T_) * HEADS_);
      attn_fused<<<ga, 256, 0, stream>>>(qkvb, biasb, attno);
    } else {
      attn_t2<<<(B_ * HW_ * HEADS_ * T_ + 255) / 256, 256, 0, stream>>>(qkvb, attno);
    }
    gemm(attno, wo, nullptr, projb, NTOK_, D_, D_, D_, D_, D_, 0);
    addv(x, projb, NTOK_ * D_);
    lnorm(x, l2g, l2b, xn, NTOK_, D_);
    gemm(xn, w1, b1, R0, NTOK_, FF_, D_, D_, FF_, FF_, 1);
    gemm(R0, w2, b2, projb, NTOK_, D_, FF_, FF_, D_, D_, 0);
    addv(x, projb, NTOK_ * D_);
  };

  // ---- encoder: spatial ----
  for (int l = 0; l < 2; ++l)
    self_layer(tok, es_ln1g + l * D_, es_ln1b + l * D_, es_wqkv + l * D_ * 3 * D_,
               es_wo + l * D_ * D_, es_ln2g + l * D_, es_ln2b + l * D_,
               es_w1 + l * D_ * FF_, es_b1 + l * FF_, es_w2 + l * FF_ * D_, es_b2 + l * D_, 1);

  // ---- encoder: temporal ----
  bt_to_tm<<<(NTOK_ * D_ / 4 + 255) / 256, 256, 0, stream>>>(tok, tmb);
  for (int l = 0; l < 2; ++l)
    self_layer(tmb, et_ln1g + l * D_, et_ln1b + l * D_, et_wqkv + l * D_ * 3 * D_,
               et_wo + l * D_ * D_, et_ln2g + l * D_, et_ln2b + l * D_,
               et_w1 + l * D_ * FF_, et_b1 + l * FF_, et_w2 + l * FF_ * D_, et_b2 + l * D_, 0);
  tm_to_bt<<<(NTOK_ * D_ / 4 + 255) / 256, 256, 0, stream>>>(tmb, tok);

  // ---- NSVQ bottleneck (tiny fp32 GEMMs) ----
  mean_k<<<(B_ * D_ / 4 + 255) / 256, 256, 0, stream>>>(tok, feat);
  gemm8(feat, vq_inw, vq_inb, zb, B_, Q_, D_, D_, Q_, Q_);
  vq_argmin<<<B_, 256, 0, stream>>>(zb, codebook, idxb);
  vq_nsvq<<<B_, 256, 0, stream>>>(zb, codebook, idxb, noise, zqb);
  gemm8(zqb, vq_outw, vq_outb, actb, B_, D_, Q_, Q_, D_, D_);

  // ---- decoder ----
  for (int l = 0; l < 2; ++l) {
    lnorm(tok, ds_ln1g + l * D_, ds_ln1b + l * D_, xn, NTOK_, D_);
    gemm(xn, ds_wqkv + l * D_ * 3 * D_, nullptr, qkvb, NTOK_, 3 * D_, D_, D_, 3 * D_, 3 * D_, 0);
    {
      dim3 ga(HW_ / AROWS, (B_ * T_) * HEADS_);
      attn_fused<<<ga, 256, 0, stream>>>(qkvb, biasb, attno);
    }
    gemm(attno, ds_wo + l * D_ * D_, nullptr, projb, NTOK_, D_, D_, D_, D_, D_, 0);
    addv(tok, projb, NTOK_ * D_);
    // cross-attn: L=1 -> softmax == 1 -> output = v @ cwo broadcast over tokens
    gemm8(actb, ds_wkv + l * D_ * 2 * D_ + D_, nullptr, v8, B_, D_, D_, D_, 2 * D_, D_);
    gemm8(v8, ds_cwo + l * D_ * D_, nullptr, vo8, B_, D_, D_, D_, D_, D_);
    bcast_add<<<(NTOK_ * D_ / 4 + 255) / 256, 256, 0, stream>>>(tok, vo8);
    // feed-forward
    lnorm(tok, ds_ln2g + l * D_, ds_ln2b + l * D_, xn, NTOK_, D_);
    gemm(xn, ds_w1 + l * D_ * FF_, ds_b1 + l * FF_, R0, NTOK_, FF_, D_, D_, FF_, FF_, 1);
    gemm(R0, ds_w2 + l * FF_ * D_, ds_b2 + l * D_, projb, NTOK_, D_, FF_, FF_, D_, D_, 0);
    addv(tok, projb, NTOK_ * D_);
  }

  // ---- pixel head ----
  gemm(tok, px_w, px_b, R0, NTOK_, CPP_, D_, D_, CPP_, CPP_, 0);
  scatter_pix<<<(out_size / 4 + 255) / 256, 256, 0, stream>>>(R0, (float*)d_out);
}
